// XNOR_Net_41575283425970
// MI455X (gfx1250) — 8-bit matrix operand order repaired and checked by reading, compiled, not run on hardware
//
#include <hip/hip_runtime.h>
#include <stdint.h>
#include <stddef.h>

typedef int v8i __attribute__((ext_vector_type(8)));
typedef int v4i __attribute__((ext_vector_type(4)));

namespace {
constexpr int kB    = 128;
constexpr int kCin  = 256;
constexpr int kH    = 56;
constexpr int kW    = 56;
constexpr int kO    = 128;
constexpr int kSP   = kH * kW;
constexpr int kK    = kCin * 9;
constexpr int kKB   = kK / 64;
constexpr int kNtot = kB * kSP;
constexpr int kNTile = 128;
constexpr int kBufB  = kNTile * 80;
}

__global__ __launch_bounds__(256) void binarize_kernel(
    const float* __restrict__ x, signed char* __restrict__ bxp,
    float* __restrict__ beta_part) {
  __shared__ __attribute__((aligned(16))) signed char tile[64][80];
  __shared__ float red[256];
  const int b   = blockIdx.z;
  const int sp0 = blockIdx.x * 64;
  const int ci0 = blockIdx.y * 64;
  const int tid = threadIdx.x;
  const int col  = tid & 63;
  const int row0 = tid >> 6;

  float asum = 0.f;
#pragma unroll
  for (int rr = 0; rr < 64; rr += 4) {
    const int ci = ci0 + row0 + rr;
    const int sp = sp0 + col;
    const float v = x[((size_t)b * kCin + ci) * kSP + sp];
    asum += fabsf(v);
    tile[col][row0 + rr] = (signed char)((v > 0.f) - (v < 0.f));
  }
  __syncthreads();

  const int orow = tid >> 2;
  const int oc   = (tid & 3) * 16;
  const v4i d = *(const v4i*)&tile[orow][oc];
  *(v4i*)(bxp + (((size_t)b * kSP + sp0 + orow) * kCin + ci0 + oc)) = d;

  red[tid] = asum;
  __syncthreads();
  for (int s = 128; s > 0; s >>= 1) {
    if (tid < s) red[tid] += red[tid + s];
    __syncthreads();
  }
  if (tid == 0) beta_part[b * 196 + blockIdx.y * 49 + blockIdx.x] = red[0];
}

__global__ __launch_bounds__(256) void beta_reduce_kernel(
    const float* __restrict__ beta_part, float* __restrict__ beta) {
  __shared__ float red[256];
  const int b = blockIdx.x, tid = threadIdx.x;
  float s = 0.f;
  if (tid < 196) s = beta_part[b * 196 + tid];
  red[tid] = s;
  __syncthreads();
  for (int st = 128; st > 0; st >>= 1) {
    if (tid < st) red[tid] += red[tid + st];
    __syncthreads();
  }
  if (tid == 0) beta[b] = red[0] * (1.f / (float)(kCin * kSP));
}

__global__ void alpha_kernel(const float* __restrict__ w, float* __restrict__ alpha) {
  const int o = blockIdx.x * blockDim.x + threadIdx.x;
  if (o < kO) {
    float s = 0.f;
    const float* p = w + (size_t)o * kK;
    for (int i = 0; i < kK; ++i) s += fabsf(p[i]);
    alpha[o] = s * (1.f / (float)kK);
  }
}

__global__ __launch_bounds__(256) void pack_weight_kernel(
    const float* __restrict__ w, signed char* __restrict__ packedA,
    signed char* __restrict__ zero_page) {
  const int kb = blockIdx.x;
  const int tid = threadIdx.x;
  if (kb == 0 && tid < 16) {
    const v4i z = {0, 0, 0, 0};
    *(v4i*)(zero_page + tid * 16) = z;
  }
  const int mb = tid >> 5, lane = tid & 31;
  const int o = mb * 16 + (lane & 15);
  const bool hi = lane >= 16;
  signed char bytes[32];
#pragma unroll
  for (int j = 0; j < 32; ++j) {
    const int v = j >> 2, c = j & 3;
    const int klocal = (v >> 2) * 32 + (v & 3) * 4 + (hi ? 16 : 0) + c;
    const int k = kb * 64 + klocal;
    const int r = k / 768, rem = k % 768;
    const int s = rem >> 8, ci = rem & 255;
    const float wv = w[(((size_t)o * kCin + ci) * 3 + r) * 3 + s];
    bytes[j] = (signed char)((wv > 0.f) - (wv < 0.f));
  }
  v4i* dst = (v4i*)(packedA + (((size_t)(kb * 8 + mb)) * 32 + lane) * 32);
  dst[0] = *(v4i*)&bytes[0];
  dst[1] = *(v4i*)&bytes[16];
}

__global__ __launch_bounds__(256) void conv_wmma_kernel(
    const signed char* __restrict__ wsbase,
    const signed char* __restrict__ packedA,
    const float* __restrict__ alpha, const float* __restrict__ beta,
    float* __restrict__ out, unsigned zero_off) {
  __shared__ __attribute__((aligned(16))) signed char btile[2 * kBufB];

  const int tid = threadIdx.x;
  const int wg = blockIdx.x;

  const int p = tid >> 1, half = tid & 1;
  const int n_st = wg * kNTile + p;
  const int b_st = n_st / kSP;
  const int sp_st = n_st - b_st * kSP;
  const int h_st = sp_st / kW;
  const int w_st = sp_st - h_st * kW;

  const int wid = tid >> 5, lane = tid & 31;
  const int wave_m = wid & 3, wave_n = wid >> 2;
  const int lane16 = lane & 15;
  const bool hi = lane >= 16;
  const int c0 = hi ? 16 : 0;

  const unsigned ldsbase = (unsigned)(size_t)(const void*)btile;
  const unsigned ldsoff_thread = (unsigned)(p * 80 + half * 32);

  auto issue_stage = [&](int kbn) {
    const int r = kbn / 12, posb = kbn - r * 12;
    const int s = posb >> 2;
    const int ciofs = (posb & 3) << 6;
    const int y = h_st + r - 1, xx = w_st + s - 1;
    unsigned voff;
    if ((unsigned)y < (unsigned)kH && (unsigned)xx < (unsigned)kW)
      voff = (unsigned)((b_st * kSP + y * kW + xx) * kCin + ciofs + half * 32);
    else
      voff = zero_off + (unsigned)(half * 32);
    const unsigned lds0 = ldsbase + (unsigned)((kbn & 1) * kBufB) + ldsoff_thread;
    asm volatile("global_load_async_to_lds_b128 %0, %1, %2\n\t"
                 "global_load_async_to_lds_b128 %0, %1, %2 offset:16"
                 :: "v"(lds0), "v"(voff), "s"(wsbase)
                 : "memory");
  };

  v8i acc[2][4] = {};

  issue_stage(0);
  for (int kb = 0; kb < kKB; ++kb) {
    asm volatile("s_wait_asynccnt 0" ::: "memory");
    __syncthreads();
    if (kb + 1 < kKB) issue_stage(kb + 1);

    const signed char* bt = btile + (kb & 1) * kBufB;

    v8i afrag[2];
#pragma unroll
    for (int i = 0; i < 2; ++i) {
      const int mb = wave_m * 2 + i;
      afrag[i] = *(const v8i*)(packedA +
          (((size_t)(kb * 8 + mb)) * 32 + lane) * 32);
    }
    if (kb + 1 < kKB) {
      __builtin_prefetch(packedA +
          (((size_t)((kb + 1) * 8 + wave_m * 2)) * 32 + lane) * 32, 0, 3);
    }

#pragma unroll
    for (int j = 0; j < 4; ++j) {
      const int nb = wave_n * 4 + j;
      const signed char* bp = bt + (nb * 16 + lane16) * 80 + c0;
      const v4i b0 = *(const v4i*)bp;
      const v4i b1 = *(const v4i*)(bp + 32);
      const v8i bfrag = {b0[0], b0[1], b0[2], b0[3], b1[0], b1[1], b1[2], b1[3]};
#pragma unroll
      for (int i = 0; i < 2; ++i) {
        acc[i][j] = __builtin_amdgcn_wmma_i32_16x16x64_iu8(
             true, afrag[i],  true, bfrag, acc[i][j],
             false,  false);
      }
    }
  }

#pragma unroll
  for (int j = 0; j < 4; ++j) {
    const int nb = wave_n * 4 + j;
    const int n = wg * kNTile + nb * 16 + lane16;
    const int b = n / kSP;
    const int sp = n - b * kSP;
    const float sc = alpha[b] * beta[b];
#pragma unroll
    for (int i = 0; i < 2; ++i) {
      const int obase = (wave_m * 2 + i) * 16 + (hi ? 8 : 0);
#pragma unroll
      for (int v = 0; v < 8; ++v) {
        const int o = obase + v;
        out[((size_t)(b * kO + o)) * kSP + sp] = (float)acc[i][j][v] * sc;
      }
    }
  }
}

extern "C" void kernel_launch(void* const* d_in, const int* in_sizes, int n_in,
                              void* d_out, int out_size, void* d_ws, size_t ws_size,
                              hipStream_t stream) {
  const float* x = (const float*)d_in[0];
  const float* w = (const float*)d_in[1];
  float* out = (float*)d_out;

  signed char* bxp = (signed char*)d_ws;
  size_t off = (size_t)kB * kSP * kCin;
  signed char* packedA = (signed char*)d_ws + off;
  off += (size_t)kKB * 8 * 32 * 32;
  float* alpha = (float*)((char*)d_ws + off); off += 512;
  float* beta  = (float*)((char*)d_ws + off); off += 512;
  float* beta_part = (float*)((char*)d_ws + off); off += (size_t)kB * 196 * 4;
  const unsigned zero_off = (unsigned)off;
  signed char* zero_page = (signed char*)d_ws + off;

  binarize_kernel<<<dim3(49, 4, kB), 256, 0, stream>>>(x, bxp, beta_part);
  beta_reduce_kernel<<<kB, 256, 0, stream>>>(beta_part, beta);
  alpha_kernel<<<1, 128, 0, stream>>>(w, alpha);
  pack_weight_kernel<<<kKB, 256, 0, stream>>>(w, packedA, zero_page);
  conv_wmma_kernel<<<kNtot / kNTile, 256, 0, stream>>>(
      (const signed char*)d_ws, packedA, alpha, beta, out, zero_off);
}
